// FrequencyWiseLSTM_75917841924613
// MI455X (gfx1250) — matrix operand order repaired and checked by reading, not compiled, not run on hardware
//
#include <hip/hip_runtime.h>
#include <math.h>

// ---------------------------------------------------------------------------
// Bidirectional 2-layer LSTM for MI455X (gfx1250, wave32, WMMA).
// B=32, T=512, IN=256, H=256 per direction, gates = 4H = 1024 (i,f,g,o).
// ---------------------------------------------------------------------------

#define B_   32
#define T_   512
#define IN_  256
#define H_   256
#define NG_  1024            // 4*H gate width
#define M_   (B_ * T_)       // 16384 rows for the input-projection GEMM
#define HS_LD 272            // padded LDS row stride for h (bf16)
#define WPAD 16              // LDS pad (elements) for staged weight rows

typedef __attribute__((ext_vector_type(16))) __bf16 v16bf;
typedef __attribute__((ext_vector_type(8)))  float  v8f;

union Frag16 { v16bf v; uint4 q[2]; };

// ---- WMMA bf16 fragment loaders (layouts per CDNA5 ISA 7.12.2) -------------
// A: 16x32 (MxK) bf16.  lane L: row M = m0 + (L&15); elements 0..7 hold
// K = kk + (L>>4)*8 .. +7 ; elements 8..15 hold K = kk + 16 + (L>>4)*8 .. +7.
__device__ inline v16bf load_a_bf16(const __bf16* base, int ld, int m0, int kk, int lane) {
    int lo = lane & 15, hi = lane >> 4;
    const __bf16* p = base + (size_t)(m0 + lo) * ld + kk + hi * 8;
    Frag16 f;
    f.q[0] = *(const uint4*)(p);        // K = kk+hi*8 .. +7      (16B)
    f.q[1] = *(const uint4*)(p + 16);   // K = kk+16+hi*8 .. +7   (16B)
    return f.v;
}
// B: 32x16 (KxN) bf16 where B[k][n] = W[n][k] (W is [N x K] row-major).
// lane L: col N = n0 + (L&15); elements 0..7 hold K = kk + (L>>4)*8 .. +7;
// elements 8..15 hold K = kk + 16 + (L>>4)*8 .. +7.
__device__ inline v16bf load_b_bf16(const __bf16* w, int ld, int n0, int kk, int lane) {
    int lo = lane & 15, hi = lane >> 4;
    const __bf16* p = w + (size_t)(n0 + lo) * ld + kk + hi * 8;
    Frag16 f;
    f.q[0] = *(const uint4*)(p);        // K = kk+hi*8 .. +7      (16B)
    f.q[1] = *(const uint4*)(p + 16);   // K = kk+16+hi*8 .. +7   (16B)
    return f.v;
}

__device__ inline v8f wmma_bf16(v16bf a, v16bf b, v8f c) {
    return __builtin_amdgcn_wmma_f32_16x16x32_bf16(false, a, false, b, (short)0, c,
                                                   false, false);
}

__device__ inline float sigmoid_f(float x) {
    return 1.0f / (1.0f + __expf(-x));
}
__device__ inline float tanh_f(float x) {
    x = fminf(fmaxf(x, -15.0f), 15.0f);
    float e = __expf(2.0f * x);
    return (e - 1.0f) / (e + 1.0f);
}

// ---------------------------------------------------------------------------
// f32 -> bf16 conversion (grid-stride)
// ---------------------------------------------------------------------------
__global__ void cvt_f32_bf16(const float* __restrict__ src, __bf16* __restrict__ dst, long n) {
    long i = (long)blockIdx.x * blockDim.x + threadIdx.x;
    long stride = (long)gridDim.x * blockDim.x;
    for (; i < n; i += stride) dst[i] = (__bf16)src[i];
}

// ---------------------------------------------------------------------------
// XG = Xb[MxK] * Wb[NxK]^T + bias   (f32 out).  Block: 256 thr = 8 waves.
// Block tile: 128(M) x 64(N).  The 64xK weight slice for this block is
// staged into LDS once (shared by all 8 waves across the whole K loop);
// B fragments then come from ds_load, A streams from global (prefetched).
// Simple loop body -> compiler picks its own unroll/pipelining within the
// register budget (manual pipelining caused spills / rotation movs).
// ---------------------------------------------------------------------------
__global__ void __launch_bounds__(256)
gemm_xg(const __bf16* __restrict__ X, const __bf16* __restrict__ W,
        const float* __restrict__ bias, float* __restrict__ Y, int K) {
    extern __shared__ char gsm[];
    __bf16* Ws = (__bf16*)gsm;               // [64][K + WPAD] bf16

    const int tid  = threadIdx.x;
    const int lane = tid & 31;
    const int wave = tid >> 5;
    const int m0 = blockIdx.x * 128 + wave * 16;
    const int nb = blockIdx.y * 64;
    const int ldw = K + WPAD;

    // ---- cooperative stage of W[nb..nb+63][0..K) into LDS (uint4 = 8 bf16) --
    {
        const int chunksPerRow = K >> 3;             // K/8 uint4 per row
        const int total = 64 * chunksPerRow;
        for (int i = tid; i < total; i += 256) {
            int r = i / chunksPerRow;
            int c = i - r * chunksPerRow;
            uint4 d = *(const uint4*)(W + (size_t)(nb + r) * K + c * 8);
            *(uint4*)(Ws + (size_t)r * ldw + c * 8) = d;
        }
    }
    __syncthreads();

    v8f zf = {0.f, 0.f, 0.f, 0.f, 0.f, 0.f, 0.f, 0.f};
    v8f acc[4] = {zf, zf, zf, zf};

    for (int kk = 0; kk < K; kk += 32) {
        // prefetch next K-slab of A into near cache (global_prefetch_b8)
        if (kk + 32 < K)
            __builtin_prefetch(X + (size_t)(m0 + (lane & 15)) * K + kk + 32, 0, 3);
        v16bf a = load_a_bf16(X, K, m0, kk, lane);       // global, streaming
#pragma unroll
        for (int j = 0; j < 4; ++j) {
            v16bf b = load_b_bf16(Ws, ldw, j * 16, kk, lane);  // LDS, hot
            acc[j] = wmma_bf16(a, b, acc[j]);
        }
    }
    const int lo = lane & 15, hi = lane >> 4;
#pragma unroll
    for (int j = 0; j < 4; ++j) {
        int col = nb + j * 16 + lo;
        float bv = bias[col];
#pragma unroll
        for (int v = 0; v < 8; ++v) {
            int row = m0 + hi * 8 + v;
            Y[(size_t)row * NG_ + col] = acc[j][v] + bv;
        }
    }
}

// ---------------------------------------------------------------------------
// Persistent bidirectional recurrence for one layer.
// grid.x = 2 chains (0 = forward, 1 = backward), block = 1024 thr = 32 waves.
// LDS: G[32][1024] f32 | C[32][256] f32 | H[32][HS_LD] bf16   (~176 KB)
// Per step: G = H @ whh^T (WMMA, A from LDS, B from L2-resident whh),
// += xg, gate nonlinearities, update H,C, stream h to out[:, t, :].
// ---------------------------------------------------------------------------
__global__ void __launch_bounds__(1024)
bilstm_recur(const float* __restrict__ xg0, const float* __restrict__ xg1,
             const __bf16* __restrict__ whh0, const __bf16* __restrict__ whh1,
             float* __restrict__ out) {
    extern __shared__ char smem[];
    float*  Gs = (float*)smem;                    // 32*1024 f32 = 128 KB
    float*  Cs = Gs + B_ * NG_;                   // 32*256  f32 =  32 KB
    __bf16* Hs = (__bf16*)(Cs + B_ * H_);         // 32*272 bf16 ≈  17 KB

    const int chain = blockIdx.x;
    const float*  xg  = chain ? xg1 : xg0;
    const __bf16* whh = chain ? whh1 : whh0;
    const int  off = chain * H_;
    const bool rev = (chain == 1);

    const int tid  = threadIdx.x;
    const int lane = tid & 31;
    const int wave = tid >> 5;
    const int m0   = (wave & 1) * 16;    // batch tile (rows 0-15 / 16-31)
    const int g4   = wave >> 1;          // 0..15 -> 64 gate columns each
    const int lo   = lane & 15, hi = lane >> 4;

    // init state
    for (int i = tid; i < B_ * H_; i += 1024) Cs[i] = 0.0f;
    for (int i = tid; i < B_ * HS_LD; i += 1024) Hs[i] = (__bf16)0.0f;
    __syncthreads();

    const v8f zf = {0.f, 0.f, 0.f, 0.f, 0.f, 0.f, 0.f, 0.f};

    for (int step = 0; step < T_; ++step) {
        const int t = rev ? (T_ - 1 - step) : step;

        // ---- G = H @ whh^T : each wave 16x64 strip, K = 256 ----
        v8f acc[4] = {zf, zf, zf, zf};
#pragma unroll 2
        for (int kk = 0; kk < H_; kk += 32) {
            v16bf a = load_a_bf16(Hs, HS_LD, m0, kk, lane);   // LDS -> ds loads
#pragma unroll
            for (int j = 0; j < 4; ++j) {
                v16bf b = load_b_bf16(whh, H_, g4 * 64 + j * 16, kk, lane); // L2-hot
                acc[j] = wmma_bf16(a, b, acc[j]);
            }
        }
#pragma unroll
        for (int j = 0; j < 4; ++j) {
            int col = g4 * 64 + j * 16 + lo;
#pragma unroll
            for (int v = 0; v < 8; ++v)
                Gs[(m0 + hi * 8 + v) * NG_ + col] = acc[j][v];
        }
        __syncthreads();

        // ---- gates + state update: 8192 (b,n) cells over 1024 threads ----
#pragma unroll
        for (int e = 0; e < 8; ++e) {
            int idx = e * 1024 + tid;
            int b = idx >> 8;
            int n = idx & 255;
            const float* xr = xg + ((size_t)b * T_ + t) * NG_;
            const float* gr = Gs + b * NG_;
            float gi = gr[n]            + xr[n];
            float gf = gr[H_ + n]       + xr[H_ + n];
            float gg = gr[2 * H_ + n]   + xr[2 * H_ + n];
            float go = gr[3 * H_ + n]   + xr[3 * H_ + n];
            float i_ = sigmoid_f(gi);
            float f_ = sigmoid_f(gf);
            float g_ = tanh_f(gg);
            float o_ = sigmoid_f(go);
            float c  = f_ * Cs[b * H_ + n] + i_ * g_;
            float h  = o_ * tanh_f(c);
            Cs[b * H_ + n] = c;
            Hs[b * HS_LD + n] = (__bf16)h;
            out[((size_t)b * T_ + t) * (2 * H_) + off + n] = h;
        }
        __syncthreads();
    }
}

// ---------------------------------------------------------------------------
// Host-side orchestration
// ---------------------------------------------------------------------------
static inline size_t align256(size_t x) { return (x + 255) & ~(size_t)255; }

extern "C" void kernel_launch(void* const* d_in, const int* in_sizes, int n_in,
                              void* d_out, int out_size, void* d_ws, size_t ws_size,
                              hipStream_t stream) {
    (void)in_sizes; (void)n_in; (void)out_size; (void)ws_size;

    // inputs (all f32)
    const float* x        = (const float*)d_in[0];
    const float* wih_l0f  = (const float*)d_in[1];
    const float* whh_l0f  = (const float*)d_in[2];
    const float* b_l0f    = (const float*)d_in[3];
    const float* wih_l0b  = (const float*)d_in[4];
    const float* whh_l0b  = (const float*)d_in[5];
    const float* b_l0b    = (const float*)d_in[6];
    const float* wih_l1f  = (const float*)d_in[7];
    const float* whh_l1f  = (const float*)d_in[8];
    const float* b_l1f    = (const float*)d_in[9];
    const float* wih_l1b  = (const float*)d_in[10];
    const float* whh_l1b  = (const float*)d_in[11];
    const float* b_l1b    = (const float*)d_in[12];
    float* outp = (float*)d_out;

    // workspace carve-up
    char* ws = (char*)d_ws;
    size_t o = 0;
    __bf16* Xb = (__bf16*)(ws + o); o = align256(o + (size_t)M_ * 512 * 2);     // 16 MB (max K)
    __bf16* wih_l0f_b = (__bf16*)(ws + o); o = align256(o + (size_t)NG_ * IN_ * 2);
    __bf16* wih_l0b_b = (__bf16*)(ws + o); o = align256(o + (size_t)NG_ * IN_ * 2);
    __bf16* whh_l0f_b = (__bf16*)(ws + o); o = align256(o + (size_t)NG_ * H_ * 2);
    __bf16* whh_l0b_b = (__bf16*)(ws + o); o = align256(o + (size_t)NG_ * H_ * 2);
    __bf16* wih_l1f_b = (__bf16*)(ws + o); o = align256(o + (size_t)NG_ * 512 * 2);
    __bf16* wih_l1b_b = (__bf16*)(ws + o); o = align256(o + (size_t)NG_ * 512 * 2);
    __bf16* whh_l1f_b = (__bf16*)(ws + o); o = align256(o + (size_t)NG_ * H_ * 2);
    __bf16* whh_l1b_b = (__bf16*)(ws + o); o = align256(o + (size_t)NG_ * H_ * 2);
    float* XG0 = (float*)(ws + o); o = align256(o + (size_t)M_ * NG_ * 4);      // 64 MB
    float* XG1 = (float*)(ws + o); o = align256(o + (size_t)M_ * NG_ * 4);      // 64 MB
    float* H1  = (float*)(ws + o); o = align256(o + (size_t)M_ * 512 * 4);      // 32 MB

    const int RSMEM = (B_ * NG_ * 4) + (B_ * H_ * 4) + (B_ * HS_LD * 2); // ~176 KB
    hipFuncSetAttribute(reinterpret_cast<const void*>(&bilstm_recur),
                        hipFuncAttributeMaxDynamicSharedMemorySize, RSMEM);
    const int GSMEM_MAX = 64 * (512 + WPAD) * 2;                         // 66 KB
    hipFuncSetAttribute(reinterpret_cast<const void*>(&gemm_xg),
                        hipFuncAttributeMaxDynamicSharedMemorySize, GSMEM_MAX);

    auto cvt = [&](const float* s, __bf16* d, long n) {
        int blocks = (int)((n + 255) / 256); if (blocks > 4096) blocks = 4096;
        cvt_f32_bf16<<<blocks, 256, 0, stream>>>(s, d, n);
    };

    // ---- convert weights once ----
    cvt(wih_l0f, wih_l0f_b, (long)NG_ * IN_);
    cvt(wih_l0b, wih_l0b_b, (long)NG_ * IN_);
    cvt(whh_l0f, whh_l0f_b, (long)NG_ * H_);
    cvt(whh_l0b, whh_l0b_b, (long)NG_ * H_);
    cvt(wih_l1f, wih_l1f_b, (long)NG_ * 512);
    cvt(wih_l1b, wih_l1b_b, (long)NG_ * 512);
    cvt(whh_l1f, whh_l1f_b, (long)NG_ * H_);
    cvt(whh_l1b, whh_l1b_b, (long)NG_ * H_);

    dim3 ggrid(M_ / 128, NG_ / 64);   // 128 x 16
    dim3 gblk(256);
    const int gsm0 = 64 * (IN_ + WPAD) * 2;   // K=256 stage
    const int gsm1 = 64 * (512 + WPAD) * 2;   // K=512 stage

    // ---- layer 0 ----
    cvt(x, Xb, (long)M_ * IN_);
    gemm_xg<<<ggrid, gblk, gsm0, stream>>>(Xb, wih_l0f_b, b_l0f, XG0, IN_);
    gemm_xg<<<ggrid, gblk, gsm0, stream>>>(Xb, wih_l0b_b, b_l0b, XG1, IN_);
    bilstm_recur<<<2, 1024, RSMEM, stream>>>(XG0, XG1, whh_l0f_b, whh_l0b_b, H1);

    // ---- layer 1 ----
    cvt(H1, Xb, (long)M_ * 512);
    gemm_xg<<<ggrid, gblk, gsm1, stream>>>(Xb, wih_l1f_b, b_l1f, XG0, 512);
    gemm_xg<<<ggrid, gblk, gsm1, stream>>>(Xb, wih_l1b_b, b_l1b, XG1, 512);
    bilstm_recur<<<2, 1024, RSMEM, stream>>>(XG0, XG1, whh_l1f_b, whh_l1b_b, outp);
}
